// AttentionGRU_21397527069132
// MI455X (gfx1250) — matrix operand order repaired and checked by reading, not compiled, not run on hardware
//
#include <hip/hip_runtime.h>
#include <hip/hip_bf16.h>
#include <stdint.h>

// ---------------------------------------------------------------------------
// AttentionGRU for MI455X (gfx1250).
// One-time f32 -> packed-bf16 conversion (memory bound), then WMMA GEMMs with
// register-double-buffered fragments (raw b128 loads, no conversion VALU).
// Each wave: 32x64 output tile = 2 A-frags x 4 B-frags -> 8 wmma per K-chunk.
// ---------------------------------------------------------------------------

typedef __attribute__((ext_vector_type(16))) __bf16 v16bf;
typedef __attribute__((ext_vector_type(8)))  float  v8f;

#define BB 128
#define TT 128
#define DD 1024
#define KP (DD / 2)            // D in packed bf16-pair (u32) units

// ---- bf16 pack helpers (only used in one-time pack + scan update) ---------
__device__ inline uint32_t f2bf(float f) {
  uint32_t u = __builtin_bit_cast(uint32_t, f);
  return (u + 0x7FFFu + ((u >> 16) & 1u)) >> 16;   // round-to-nearest-even
}
__device__ inline uint32_t pack2(float lo, float hi) {
  return f2bf(lo) | (f2bf(hi) << 16);
}

__device__ inline v8f wmma_bf16(v16bf a, v16bf b, v8f c) {
  // emits v_wmma_f32_16x16x32_bf16 : D = A(16x32)*B(32x16) + C
  return __builtin_amdgcn_wmma_f32_16x16x32_bf16(
      false, a, false, b, (short)0, c, false, false);
}

// ---- fragment loads from packed-pair arrays (wave32, ISA 7.12.2 layouts) --
struct Frag32 { uint4 lo; uint4 hi; };   // 32 bytes == v16bf

// A (16x32, MxK): lane<16 -> M=lane, K {0..7,16..23}; lane>=16 -> M=lane-16,
// K {8..15,24..31}. Packed u32 j holds K pair (2j,2j+1).
__device__ inline v16bf loadA_packed(const uint32_t* __restrict__ Xp,
                                     size_t ldp, int m0, int kp) {
  int lane = threadIdx.x & 31;
  int half = lane >> 4;
  int m    = lane & 15;
  const uint32_t* row = Xp + (size_t)(m0 + m) * ldp + kp;
  Frag32 f;
  f.lo = *(const uint4*)(row + half * 4);
  f.hi = *(const uint4*)(row + 8 + half * 4);
  return __builtin_bit_cast(v16bf, f);
}

// B (32x16, KxN) with B[k][n] = Wt[n][k], Wt packed row-major in k-pairs:
// lane<16 -> N=lane, K {0..7,16..23}; lane>=16 -> N=lane-16, K {8..15,24..31}.
__device__ inline v16bf loadB_packed(const uint32_t* __restrict__ Wp,
                                     size_t ldp, int n0, int kp) {
  int lane = threadIdx.x & 31;
  int half = lane >> 4;
  int n    = lane & 15;
  const uint32_t* row = Wp + (size_t)(n0 + n) * ldp + kp;
  Frag32 f;
  f.lo = *(const uint4*)(row + half * 4);
  f.hi = *(const uint4*)(row + 8 + half * 4);
  return __builtin_bit_cast(v16bf, f);
}

// ---- C/D store (f32 16x16: VGPR i -> M=i (lane<16) / M=i+8 (lane>=16)) ----
__device__ inline void storeC(float* __restrict__ out, size_t ld, int m0,
                              int n0, v8f c) {
  int lane = threadIdx.x & 31;
  int half = lane >> 4;
  int col  = n0 + (lane & 15);
#pragma unroll
  for (int i = 0; i < 8; ++i) {
    out[(size_t)(m0 + i + half * 8) * ld + col] = c[i];
  }
}

// ---- 32x64 wave tile, register-double-buffered over K ---------------------
__device__ inline void gemm_wave_tile(const uint32_t* __restrict__ Xp,
                                      size_t ldx, int m0,
                                      const uint32_t* __restrict__ Wp,
                                      size_t ldw, int n0, int Kp,
                                      v8f acc[2][4]) {
  v16bf a0 = loadA_packed(Xp, ldx, m0, 0);
  v16bf a1 = loadA_packed(Xp, ldx, m0 + 16, 0);
  v16bf b0 = loadB_packed(Wp, ldw, n0, 0);
  v16bf b1 = loadB_packed(Wp, ldw, n0 + 16, 0);
  v16bf b2 = loadB_packed(Wp, ldw, n0 + 32, 0);
  v16bf b3 = loadB_packed(Wp, ldw, n0 + 48, 0);
  for (int kp = 0; kp < Kp; kp += 16) {
    v16bf ca0 = a0, ca1 = a1, cb0 = b0, cb1 = b1, cb2 = b2, cb3 = b3;
    const int kn = kp + 16;
    if (kn < Kp) {
      __builtin_prefetch(Xp + (size_t)m0 * ldx + kn + 16, 0, 1);
      a0 = loadA_packed(Xp, ldx, m0, kn);
      a1 = loadA_packed(Xp, ldx, m0 + 16, kn);
      b0 = loadB_packed(Wp, ldw, n0, kn);
      b1 = loadB_packed(Wp, ldw, n0 + 16, kn);
      b2 = loadB_packed(Wp, ldw, n0 + 32, kn);
      b3 = loadB_packed(Wp, ldw, n0 + 48, kn);
    }
    acc[0][0] = wmma_bf16(ca0, cb0, acc[0][0]);
    acc[1][0] = wmma_bf16(ca1, cb0, acc[1][0]);
    acc[0][1] = wmma_bf16(ca0, cb1, acc[0][1]);
    acc[1][1] = wmma_bf16(ca1, cb1, acc[1][1]);
    acc[0][2] = wmma_bf16(ca0, cb2, acc[0][2]);
    acc[1][2] = wmma_bf16(ca1, cb2, acc[1][2]);
    acc[0][3] = wmma_bf16(ca0, cb3, acc[0][3]);
    acc[1][3] = wmma_bf16(ca1, cb3, acc[1][3]);
  }
}

// ===========================================================================
// Pack kernel: f32 -> packed bf16 pairs (u32). One float4 -> uint2 per thread.
// ===========================================================================
__global__ __launch_bounds__(256) void pack_bf16(
    const float4* __restrict__ src, uint2* __restrict__ dst, int n4) {
  int i = blockIdx.x * 256 + threadIdx.x;
  if (i < n4) {
    float4 v = src[i];
    uint2 r;
    r.x = pack2(v.x, v.y);
    r.y = pack2(v.z, v.w);
    dst[i] = r;
  }
}

// ===========================================================================
// GEMM: Out[m,n] = sum_d X[m,d] * Wt[n,d], X/Wt pre-packed bf16 pairs.
// Block 256 = 8 waves, each wave a 32(M) x 64(N) tile; block tile 256x64.
// ===========================================================================
__global__ __launch_bounds__(256) void gemm_bf16(
    const uint32_t* __restrict__ Xp, const uint32_t* __restrict__ Wp,
    float* __restrict__ Out, int N, int Kp) {
  const int m0 = blockIdx.x * 256 + (threadIdx.x >> 5) * 32;
  const int n0 = blockIdx.y * 64;
  v8f acc[2][4] = {};
  gemm_wave_tile(Xp, Kp, m0, Wp, Kp, n0, Kp, acc);
#pragma unroll
  for (int r = 0; r < 2; ++r)
#pragma unroll
    for (int j = 0; j < 4; ++j)
      storeC(Out, N, m0 + r * 16, n0 + j * 16, acc[r][j]);
}

// ===========================================================================
// Persistent GRU scan: 16 blocks x 8 waves = 128 waves covering the
// (2 GEMMs) x (4 M-tiles of 32) x (16 N-tiles of 64) per-step work exactly.
// ===========================================================================
#define SCAN_BLOCKS 16

__device__ inline void grid_barrier(unsigned* cnt, unsigned* target) {
  __syncthreads();
  *target += SCAN_BLOCKS;
  if (threadIdx.x == 0) {
    __hip_atomic_fetch_add(cnt, 1u, __ATOMIC_ACQ_REL, __HIP_MEMORY_SCOPE_AGENT);
    while (__hip_atomic_load(cnt, __ATOMIC_ACQUIRE, __HIP_MEMORY_SCOPE_AGENT) <
           *target) {
      __builtin_amdgcn_s_sleep(2);
    }
  }
  __syncthreads();
}

__global__ __launch_bounds__(256) void gru_scan(
    const float* __restrict__ fWr, const float* __restrict__ fW,
    const float* __restrict__ g,   const uint32_t* __restrict__ UrwP,
    const float* __restrict__ Urb, const uint32_t* __restrict__ UwP,
    const float* __restrict__ Ub,  const int* __restrict__ num_facts,
    float* hA, float* hB, uint32_t* hPA, uint32_t* hPB,
    float* aR, float* aW, unsigned* sync_cnt, float* __restrict__ out) {
  const int wave_g = blockIdx.x * 8 + (threadIdx.x >> 5);  // 0..127
  const int gsel   = wave_g >> 6;                          // 0: r-GEMM, 1: w
  const int tile   = wave_g & 63;
  const int mt     = tile >> 4;                            // rows mt*32 (0..3)
  const int nt     = tile & 15;                            // cols nt*64
  const uint32_t* Wsel = gsel ? UwP : UrwP;
  float*          Osel = gsel ? aW : aR;

  const int tid   = blockIdx.x * 256 + threadIdx.x;        // 0..4095
  const int base  = tid * 32;                              // 32 consecutive
  const int bidx  = base >> 10;
  const int t_out = num_facts[bidx] - 1;

  unsigned target = 0;
  for (int t = 0; t < TT; ++t) {
    const float*    hc  = (t & 1) ? hB : hA;
    float*          hn  = (t & 1) ? hA : hB;
    const uint32_t* hPc = (t & 1) ? hPB : hPA;
    uint32_t*       hPn = (t & 1) ? hPA : hPB;

    // ---- GEMM: Osel[m,n] = sum_d h[m,d] * Wsel[n,d] -----------------------
    v8f acc[2][4] = {};
    gemm_wave_tile(hPc, KP, mt * 32, Wsel, KP, nt * 64, KP, acc);
#pragma unroll
    for (int r = 0; r < 2; ++r)
#pragma unroll
      for (int j = 0; j < 4; ++j)
        storeC(Osel, DD, mt * 32 + r * 16, nt * 64 + j * 16, acc[r][j]);

    __threadfence();
    grid_barrier(sync_cnt, &target);

    // ---- elementwise GRU update: 32 consecutive elems per thread ----------
    {
      const float gt = g[bidx * TT + t];
      const float om = 1.0f - gt;
#pragma unroll
      for (int i = 0; i < 8; ++i) {
        int e = base + i * 4;
        int k = e & (DD - 1);
        size_t ft = ((size_t)bidx * TT + t) * DD + k;
        float4 fr  = *(const float4*)(fWr + ft);
        float4 fw  = *(const float4*)(fW + ft);
        float4 ar  = *(const float4*)(aR + e);
        float4 aw  = *(const float4*)(aW + e);
        float4 br  = *(const float4*)(Urb + k);
        float4 bw  = *(const float4*)(Ub + k);
        float4 hcv = *(const float4*)(hc + e);
        float4 hv;
        {
          float r0 = 1.0f / (1.0f + __expf(-(fr.x + ar.x + br.x)));
          float r1 = 1.0f / (1.0f + __expf(-(fr.y + ar.y + br.y)));
          float r2 = 1.0f / (1.0f + __expf(-(fr.z + ar.z + br.z)));
          float r3 = 1.0f / (1.0f + __expf(-(fr.w + ar.w + br.w)));
          hv.x = gt * tanhf(fw.x + r0 * (aw.x + bw.x)) + om * hcv.x;
          hv.y = gt * tanhf(fw.y + r1 * (aw.y + bw.y)) + om * hcv.y;
          hv.z = gt * tanhf(fw.z + r2 * (aw.z + bw.z)) + om * hcv.z;
          hv.w = gt * tanhf(fw.w + r3 * (aw.w + bw.w)) + om * hcv.w;
        }
        *(float4*)(hn + e) = hv;
        uint2 p;
        p.x = pack2(hv.x, hv.y);
        p.y = pack2(hv.z, hv.w);
        *(uint2*)(hPn + (e >> 1)) = p;
        if (t_out == t) *(float4*)(out + e) = hv;
      }
    }

    __threadfence();
    grid_barrier(sync_cnt, &target);
  }
}

// ===========================================================================
extern "C" void kernel_launch(void* const* d_in, const int* in_sizes, int n_in,
                              void* d_out, int out_size, void* d_ws,
                              size_t ws_size, hipStream_t stream) {
  (void)in_sizes; (void)n_in; (void)out_size; (void)ws_size;
  const float* facts   = (const float*)d_in[0];   // (B,T,D)
  const int*   numf    = (const int*)  d_in[1];   // (B,)
  const float* g       = (const float*)d_in[2];   // (B,T,1)
  const float* mem_old = (const float*)d_in[3];   // (B,1,D)
  const float* Wr      = (const float*)d_in[4];   // (D,D)
  const float* Urw     = (const float*)d_in[5];   // (D,D)
  const float* Urb     = (const float*)d_in[6];   // (D,)
  const float* W       = (const float*)d_in[7];   // (D,D)
  const float* Uw      = (const float*)d_in[8];   // (D,D)
  const float* Ub      = (const float*)d_in[9];   // (D,)
  float* out = (float*)d_out;                     // (B,D)

  const size_t nBTD = (size_t)BB * TT * DD;       // 16M
  const size_t nBD  = (size_t)BB * DD;            // 128K
  const size_t nDD  = (size_t)DD * DD;            // 1M

  // Workspace layout
  float*    fWr    = (float*)d_ws;                // 64 MB
  float*    fW     = fWr + nBTD;                  // 64 MB
  uint32_t* factsP = (uint32_t*)(fW + nBTD);      // 32 MB
  uint32_t* WrP    = factsP + nBTD / 2;           // 2 MB each
  uint32_t* WP     = WrP + nDD / 2;
  uint32_t* UrwP   = WP + nDD / 2;
  uint32_t* UwP    = UrwP + nDD / 2;
  float*    hA     = (float*)(UwP + nDD / 2);
  float*    hB     = hA + nBD;
  uint32_t* hPA    = (uint32_t*)(hB + nBD);
  uint32_t* hPB    = hPA + nBD / 2;
  float*    aR     = (float*)(hPB + nBD / 2);
  float*    aW     = aR + nBD;
  unsigned* sync_cnt = (unsigned*)(aW + nBD);

  hipMemsetAsync(sync_cnt, 0, 256, stream);
  hipMemcpyAsync(hA, mem_old, nBD * sizeof(float), hipMemcpyDeviceToDevice,
                 stream);

  // One-time f32 -> packed bf16 conversions (memory bound)
  pack_bf16<<<(int)(nBTD / 4 / 256), 256, 0, stream>>>((const float4*)facts,
                                                       (uint2*)factsP,
                                                       (int)(nBTD / 4));
  pack_bf16<<<(int)(nDD / 4 / 256), 256, 0, stream>>>((const float4*)Wr,
                                                      (uint2*)WrP,
                                                      (int)(nDD / 4));
  pack_bf16<<<(int)(nDD / 4 / 256), 256, 0, stream>>>((const float4*)W,
                                                      (uint2*)WP,
                                                      (int)(nDD / 4));
  pack_bf16<<<(int)(nDD / 4 / 256), 256, 0, stream>>>((const float4*)Urw,
                                                      (uint2*)UrwP,
                                                      (int)(nDD / 4));
  pack_bf16<<<(int)(nDD / 4 / 256), 256, 0, stream>>>((const float4*)Uw,
                                                      (uint2*)UwP,
                                                      (int)(nDD / 4));
  pack_bf16<<<(int)(nBD / 4 / 256), 256, 0, stream>>>((const float4*)mem_old,
                                                      (uint2*)hPA,
                                                      (int)(nBD / 4));

  // Phase 1: fWr = facts . Wr^T ; fW = facts . W^T  (M=16384, N=1024, K=1024)
  dim3 grid1((BB * TT) / 256, DD / 64, 1);
  gemm_bf16<<<grid1, 256, 0, stream>>>(factsP, WrP, fWr, DD, KP);
  gemm_bf16<<<grid1, 256, 0, stream>>>(factsP, WP,  fW,  DD, KP);

  // Phase 2: persistent cooperative scan (2 GEMMs + gate update per step)
  gru_scan<<<SCAN_BLOCKS, 256, 0, stream>>>(fWr, fW, g, UrwP, Urb, UwP, Ub,
                                            numf, hA, hB, hPA, hPB, aR, aW,
                                            sync_cnt, out);
}
